// SwinTransformer3D_73280732004620
// MI455X (gfx1250) — matrix operand order repaired and checked by reading, not compiled, not run on hardware
//
#include <hip/hip_runtime.h>
#include <hip/hip_bf16.h>
#include <cmath>

typedef _Float16 half_t;
typedef __attribute__((ext_vector_type(16))) _Float16 v16h;
typedef __attribute__((ext_vector_type(8)))  float    v8f;

// ---------------------------------------------------------------------------
// WMMA helpers (gfx1250, wave32).  Fragment layouts used here (contiguous K order, same for A and B):
//  A (16x32 f16): lane l -> row = l&15, K-halfgroup g=l>>4; reg j holds
//                 K = g*16+j
//  B (32x16 f16): lane l -> col = l&15, g=l>>4; reg j holds K = g*16+j
//  C/D (16x16 f32): lane l -> col = l&15, g=l>>4; reg r holds row r+8*g
// ---------------------------------------------------------------------------
__device__ __forceinline__ v8f zero8() {
  v8f z;
#pragma unroll
  for (int i = 0; i < 8; ++i) z[i] = 0.0f;
  return z;
}

__device__ __forceinline__ v8f wmma_f16(v16h a, v16h b, v8f c) {
  return __builtin_amdgcn_wmma_f32_16x16x32_f16(false, a, false, b, (short)0, c,
                                                false, false);
}

// A fragment from row-major LDS tile (stride in halves), rows row0..row0+15,
// K columns k0..k0+31
__device__ __forceinline__ v16h frag_a_lds(const half_t* s, int row0, int stride,
                                           int k0, int lane) {
  const half_t* p = s + (row0 + (lane & 15)) * stride + k0 + ((lane >> 4) << 4);
  v16h a;
#pragma unroll
  for (int j = 0; j < 8; ++j) a[j] = p[j];
#pragma unroll
  for (int j = 0; j < 8; ++j) a[8 + j] = p[8 + j];
  return a;
}

// B fragment: B[k][n] = s[(k0+k)*stride + col0 + n]
__device__ __forceinline__ v16h frag_b_lds(const half_t* s, int k0, int col0,
                                           int stride, int lane) {
  const half_t* p = s + (k0 + ((lane >> 4) << 4)) * stride + col0 + (lane & 15);
  v16h b;
#pragma unroll
  for (int j = 0; j < 16; ++j) b[j] = p[j * stride];
  return b;
}

// B fragment, transposed source: B[k][n] = s[(row0+n)*stride + k]  (for q.k^T)
__device__ __forceinline__ v16h frag_bT_lds(const half_t* s, int row0, int stride,
                                            int lane) {
  const half_t* p = s + (row0 + (lane & 15)) * stride + ((lane >> 4) << 4);
  v16h b;
#pragma unroll
  for (int j = 0; j < 16; ++j) b[j] = p[j];
  return b;
}

__device__ __forceinline__ float wred_sum(float v) {
#pragma unroll
  for (int m = 16; m > 0; m >>= 1) v += __shfl_xor(v, m, 32);
  return v;
}
__device__ __forceinline__ float wred_max(float v) {
#pragma unroll
  for (int m = 16; m > 0; m >>= 1) v = fmaxf(v, __shfl_xor(v, m, 32));
  return v;
}

// ---------------------------------------------------------------------------
// f32 -> f16 weight conversion
// ---------------------------------------------------------------------------
__global__ __launch_bounds__(256) void f2h_kernel(const float* __restrict__ s,
                                                  half_t* __restrict__ d, int n) {
  int i = blockIdx.x * 256 + threadIdx.x;
  if (i < n) d[i] = (half_t)s[i];
}

// ---------------------------------------------------------------------------
// adaLN modulation:  mod = silu(t) @ ada_w1 @ ada_w2 + ada_b2   (B=2)
// ---------------------------------------------------------------------------
__global__ __launch_bounds__(256) void mod_kernel(const float* __restrict__ t,
                                                  const float* __restrict__ w1,
                                                  const float* __restrict__ w2,
                                                  const float* __restrict__ b2,
                                                  float* __restrict__ mod) {
  __shared__ float u[2 * 48];
  const int tid = threadIdx.x;
  if (tid < 96) {
    int b = tid / 48, r = tid % 48;
    float s = 0.f;
    for (int k = 0; k < 384; ++k) {
      float tv = t[b * 384 + k];
      s += (tv / (1.f + __expf(-tv))) * w1[k * 48 + r];
    }
    u[tid] = s;
  }
  __syncthreads();
  for (int o = tid; o < 2 * 2304; o += 256) {
    int b = o / 2304, c = o - b * 2304;
    float s = b2[c];
    for (int k = 0; k < 48; ++k) s += u[b * 48 + k] * w2[k * 2304 + c];
    mod[o] = s;
  }
}

// ---------------------------------------------------------------------------
// LN + modulate + shifted roll + window partition.  One wave per padded token
// row (512 windows * 128 rows).  Pad rows (t>=98) are zero-filled.
// ---------------------------------------------------------------------------
__global__ __launch_bounds__(256) void ln_part_kernel(const float* __restrict__ x,
                                                      const float* __restrict__ mod,
                                                      half_t* __restrict__ xw) {
  const int lane = threadIdx.x & 31;
  const int gw = blockIdx.x * 8 + (threadIdx.x >> 5);
  const int widx = gw >> 7;
  const int t = gw & 127;
  half_t* dst = xw + (size_t)gw * 384;
  if (t >= 98) {
#pragma unroll
    for (int j = 0; j < 12; ++j) dst[lane + 32 * j] = (half_t)0.f;
    return;
  }
  const int b = widx >> 8, dwin = (widx >> 6) & 3, hwin = (widx >> 3) & 7,
            wwin = widx & 7;
  const int dt = t / 49, rem = t - dt * 49, ht = rem / 7, wt = rem - ht * 7;
  int ds = dwin * 2 + dt + 1;  if (ds >= 8)  ds -= 8;
  int hs = hwin * 7 + ht + 3;  if (hs >= 56) hs -= 56;
  int wsp = wwin * 7 + wt + 3; if (wsp >= 56) wsp -= 56;
  const float* src = x + (((size_t)(b * 8 + ds) * 56 + hs) * 56 + wsp) * 384;
  float v[12]; float s = 0.f;
#pragma unroll
  for (int j = 0; j < 12; ++j) { v[j] = src[lane + 32 * j]; s += v[j]; }
  s = wred_sum(s);
  const float mean = s * (1.f / 384.f);
  float s2 = 0.f;
#pragma unroll
  for (int j = 0; j < 12; ++j) { float d = v[j] - mean; s2 += d * d; }
  s2 = wred_sum(s2);
  const float rstd = rsqrtf(s2 * (1.f / 384.f) + 1e-6f);
  const float* mb = mod + b * 2304;  // [shift | scale | gate | ...]
#pragma unroll
  for (int j = 0; j < 12; ++j) {
    int c = lane + 32 * j;
    float h = (v[j] - mean) * rstd;
    dst[c] = (half_t)(h * (1.f + mb[384 + c]) + mb[c]);
  }
}

// ---------------------------------------------------------------------------
// LN + modulate for MLP branch (no shift/partition)
// ---------------------------------------------------------------------------
__global__ __launch_bounds__(256) void ln_mlp_kernel(const float* __restrict__ x1,
                                                     const float* __restrict__ mod,
                                                     half_t* __restrict__ h2) {
  const int lane = threadIdx.x & 31;
  const int row = blockIdx.x * 8 + (threadIdx.x >> 5);  // 0..50175
  const int b = row / 25088;
  const float* src = x1 + (size_t)row * 384;
  float v[12]; float s = 0.f;
#pragma unroll
  for (int j = 0; j < 12; ++j) { v[j] = src[lane + 32 * j]; s += v[j]; }
  s = wred_sum(s);
  const float mean = s * (1.f / 384.f);
  float s2 = 0.f;
#pragma unroll
  for (int j = 0; j < 12; ++j) { float d = v[j] - mean; s2 += d * d; }
  s2 = wred_sum(s2);
  const float rstd = rsqrtf(s2 * (1.f / 384.f) + 1e-6f);
  const float* mb = mod + b * 2304;
  half_t* dst = h2 + (size_t)row * 384;
#pragma unroll
  for (int j = 0; j < 12; ++j) {
    int c = lane + 32 * j;
    float h = (v[j] - mean) * rstd;
    dst[c] = (half_t)(h * (1.f + mb[1536 + c]) + mb[1152 + c]);
  }
}

// ---------------------------------------------------------------------------
// Generic WMMA GEMM: C[M,Nn] = A[M,K](f16) @ B[K,Nn](f16) + bias, 4 epilogues.
// BM=128, BN=64, BK=32, 256 threads (8 waves, one 16-row strip each).
// EPI: 0=f16 out, 1=gelu->f16, 2=f32 out, 3=fused final residual (float out)
// ---------------------------------------------------------------------------
template <int EPI>
__global__ __launch_bounds__(256) void gemm_kernel(
    const half_t* __restrict__ A, const half_t* __restrict__ B,
    const float* __restrict__ bias, void* __restrict__ Cout, int M, int Nn,
    int K, const float* __restrict__ resid, const float* __restrict__ modp,
    int rows_per_b) {
  __shared__ __align__(16) half_t As[128 * 40];
  __shared__ __align__(16) half_t Bs[32 * 72];
  const int tid = threadIdx.x;
  const int lane = tid & 31;
  const int wv = tid >> 5;
  const int bm = blockIdx.x * 128;
  const int bn = blockIdx.y * 64;

  v8f acc0 = zero8(), acc1 = zero8(), acc2 = zero8(), acc3 = zero8();

  const int ar = tid >> 1, aoff = (tid & 1) * 16;   // A: 128 rows x 32 halves
  const int br = tid >> 3, bc = (tid & 7) * 8;      // B: 32 rows x 64 halves

  for (int k0 = 0; k0 < K; k0 += 32) {
    const float4* ap = (const float4*)(A + (size_t)(bm + ar) * K + k0 + aoff);
    float4 a0 = ap[0], a1 = ap[1];
    float4 b0 = *(const float4*)(B + (size_t)(k0 + br) * Nn + bn + bc);
    if (k0 + 32 < K)
      __builtin_prefetch(A + (size_t)(bm + ar) * K + k0 + 32 + aoff, 0, 1);
    *(float4*)&As[ar * 40 + aoff] = a0;
    *(float4*)&As[ar * 40 + aoff + 8] = a1;
    *(float4*)&Bs[br * 72 + bc] = b0;
    __syncthreads();
    v16h af = frag_a_lds(As, wv * 16, 40, 0, lane);
    v16h bf0 = frag_b_lds(Bs, 0, 0, 72, lane);
    v16h bf1 = frag_b_lds(Bs, 0, 16, 72, lane);
    v16h bf2 = frag_b_lds(Bs, 0, 32, 72, lane);
    v16h bf3 = frag_b_lds(Bs, 0, 48, 72, lane);
    acc0 = wmma_f16(af, bf0, acc0);
    acc1 = wmma_f16(af, bf1, acc1);
    acc2 = wmma_f16(af, bf2, acc2);
    acc3 = wmma_f16(af, bf3, acc3);
    __syncthreads();
  }
  const int n = lane & 15, g = lane >> 4;
  v8f accs[4] = {acc0, acc1, acc2, acc3};
#pragma unroll
  for (int nt = 0; nt < 4; ++nt) {
    const int col = bn + nt * 16 + n;
    const float bval = bias[col];
#pragma unroll
    for (int r = 0; r < 8; ++r) {
      const int row = bm + wv * 16 + g * 8 + r;
      float v = accs[nt][r] + bval;
      if constexpr (EPI == 0) {
        ((half_t*)Cout)[(size_t)row * Nn + col] = (half_t)v;
      } else if constexpr (EPI == 1) {
        float ge = 0.5f * v * (1.f + erff(v * 0.70710678118f));
        ((half_t*)Cout)[(size_t)row * Nn + col] = (half_t)ge;
      } else if constexpr (EPI == 2) {
        ((float*)Cout)[(size_t)row * Nn + col] = v;
      } else {
        int b = row / rows_per_b;
        float gm = modp[b * 2304 + 1920 + col];  // g_mlp
        ((float*)Cout)[(size_t)row * Nn + col] =
            resid[(size_t)row * Nn + col] + gm * v;
      }
    }
  }
}

// ---------------------------------------------------------------------------
// Windowed attention: one block per (window, head, 64-row half).
// qkv layout: [512*128, 1152] f16, col = sel*384 + head*32 + dim.
// rel-pos bias and shift mask computed analytically (host constants in ref).
// ---------------------------------------------------------------------------
__global__ __launch_bounds__(256) void attn_kernel(const half_t* __restrict__ qkv,
                                                   const float* __restrict__ rel_bias,
                                                   half_t* __restrict__ out) {
  __shared__ __align__(16) half_t k_s[128 * 40];
  __shared__ __align__(16) half_t v_s[128 * 40];
  __shared__ __align__(16) half_t q_s[64 * 40];
  __shared__ float S[64 * 128];  // scores; rows reused in-place as f16 probs

  const int tid = threadIdx.x;
  const int lane = tid & 31;
  const int wv = tid >> 5;
  const int widx = blockIdx.x / 24;
  const int rem = blockIdx.x % 24;
  const int head = rem >> 1;
  const int part = rem & 1;

  // ---- stage K,V (all 128 rows) and Q (this 64-row half) into LDS ----
  {
    const int r = tid >> 1, off = (tid & 1) * 16;
    const size_t base = ((size_t)widx * 128 + r) * 1152 + head * 32 + off;
    const float4* kp = (const float4*)(qkv + base + 384);
    const float4* vp = (const float4*)(qkv + base + 768);
    float4 ka = kp[0], kb = kp[1];
    float4 va = vp[0], vb = vp[1];
    *(float4*)&k_s[r * 40 + off] = ka;
    *(float4*)&k_s[r * 40 + off + 8] = kb;
    *(float4*)&v_s[r * 40 + off] = va;
    *(float4*)&v_s[r * 40 + off + 8] = vb;
    if (tid < 128) {
      const size_t qb =
          ((size_t)widx * 128 + part * 64 + r) * 1152 + head * 32 + off;
      const float4* qp = (const float4*)(qkv + qb);
      float4 qa = qp[0], qv = qp[1];
      *(float4*)&q_s[r * 40 + off] = qa;
      *(float4*)&q_s[r * 40 + off + 8] = qv;
    }
  }
  __syncthreads();

  // ---- S = Q @ K^T : 4 M-tiles x 8 J-tiles, wave = (mtile, j-half) ----
  {
    const int mt = wv >> 1;
    const int jh = wv & 1;
    v16h a = frag_a_lds(q_s, mt * 16, 40, 0, lane);
    const int n = lane & 15, g = lane >> 4;
#pragma unroll
    for (int jt = jh * 4; jt < jh * 4 + 4; ++jt) {
      v16h b = frag_bT_lds(k_s, jt * 16, 40, lane);
      v8f c = wmma_f16(a, b, zero8());
#pragma unroll
      for (int r = 0; r < 8; ++r)
        S[(mt * 16 + g * 8 + r) * 128 + jt * 16 + n] = c[r];
    }
  }
  __syncthreads();

  // ---- scale + rel bias + shift mask + softmax, 8 rows per wave ----
  {
    const int dwin = (widx >> 6) & 3, hwin = (widx >> 3) & 7, wwin = widx & 7;
    const float qscale = 0.17677669529663687f;  // HD^-0.5
    for (int lr = 0; lr < 8; ++lr) {
      const int il = wv * 8 + lr;
      const int i = part * 64 + il;
      const int dti = i / 49, remi = i - dti * 49, hti = remi / 7,
                wti = remi - hti * 7;
      const int dpi = dwin * 2 + dti, hpi = hwin * 7 + hti, wpi = wwin * 7 + wti;
      const int li = (dpi < 6 ? 0 : (dpi < 7 ? 1 : 2)) * 6 +
                     (hpi < 49 ? 0 : (hpi < 53 ? 1 : 2)) * 2 + (wpi < 49 ? 0 : 1);
      float val[4];
#pragma unroll
      for (int jj = 0; jj < 4; ++jj) {
        const int j = lane + jj * 32;
        float sv = S[il * 128 + j] * qscale;
        if (j >= 98) {
          val[jj] = -1e9f;  // pad column
        } else if (i < 98) {
          const int dtj = j / 49, remj = j - dtj * 49, htj = remj / 7,
                    wtj = remj - htj * 7;
          const int dpj = dwin * 2 + dtj, hpj = hwin * 7 + htj,
                    wpj = wwin * 7 + wtj;
          const int lj = (dpj < 6 ? 0 : (dpj < 7 ? 1 : 2)) * 6 +
                         (hpj < 49 ? 0 : (hpj < 53 ? 1 : 2)) * 2 +
                         (wpj < 49 ? 0 : 1);
          const int idx =
              (dti - dtj + 1) * 169 + (hti - htj + 6) * 13 + (wti - wtj + 6);
          sv += rel_bias[idx * 12 + head];
          if (li != lj) sv -= 100.0f;
          val[jj] = sv;
        } else {
          val[jj] = sv;  // pad row: keep softmax well-defined
        }
      }
      float mx = fmaxf(fmaxf(val[0], val[1]), fmaxf(val[2], val[3]));
      mx = wred_max(mx);
      float e[4], ssum = 0.f;
#pragma unroll
      for (int jj = 0; jj < 4; ++jj) {
        e[jj] = __expf(val[jj] - mx);
        ssum += e[jj];
      }
      ssum = wred_sum(ssum);
      const float rinv = 1.0f / ssum;
      half_t* Prow = (half_t*)&S[il * 128];  // in-place f16 probs
#pragma unroll
      for (int jj = 0; jj < 4; ++jj)
        Prow[lane + jj * 32] = (half_t)(e[jj] * rinv);
    }
  }
  __syncthreads();

  // ---- O = P @ V : 4 M-tiles x 2 dim-tiles, one per wave, K loop of 4 ----
  {
    const int mt = wv >> 1;
    const int nt = wv & 1;
    v8f c = zero8();
#pragma unroll
    for (int kk = 0; kk < 4; ++kk) {
      v16h a = frag_a_lds((const half_t*)S, mt * 16, 256, kk * 32, lane);
      v16h b = frag_b_lds(v_s, kk * 32, nt * 16, 40, lane);
      c = wmma_f16(a, b, c);
    }
    const int n = lane & 15, g = lane >> 4;
#pragma unroll
    for (int r = 0; r < 8; ++r) {
      const int row = part * 64 + mt * 16 + g * 8 + r;
      out[((size_t)widx * 128 + row) * 384 + head * 32 + nt * 16 + n] =
          (half_t)c[r];
    }
  }
}

// ---------------------------------------------------------------------------
// Reverse-roll / window-unpartition + gated residual:
//   x1 = x + g_msa * proj_out(window token of rolled coords)
// ---------------------------------------------------------------------------
__global__ __launch_bounds__(256) void attn_residual_kernel(
    const float* __restrict__ x, const float* __restrict__ mod,
    const float* __restrict__ proj, float* __restrict__ out) {
  const size_t i = (size_t)blockIdx.x * 256 + threadIdx.x;
  const int c = (int)(i % 384);
  const size_t tok = i / 384;
  const int w = (int)(tok % 56);
  const size_t t2 = tok / 56;
  const int h = (int)(t2 % 56);
  const size_t t3 = t2 / 56;
  const int d = (int)(t3 % 8);
  const int b = (int)(t3 / 8);
  const int dp = (d + 7) & 7;
  int hp = h - 3;  if (hp < 0) hp += 56;
  int wp = w - 3;  if (wp < 0) wp += 56;
  const int dwin = dp >> 1, dt = dp & 1;
  const int hwin = hp / 7, ht = hp - hwin * 7;
  const int wwin = wp / 7, wt = wp - wwin * 7;
  const int widx = ((b * 4 + dwin) * 8 + hwin) * 8 + wwin;
  const int tkn = dt * 49 + ht * 7 + wt;
  out[i] = x[i] + mod[b * 2304 + 768 + c] * proj[((size_t)widx * 128 + tkn) * 384 + c];
}

// ---------------------------------------------------------------------------
extern "C" void kernel_launch(void* const* d_in, const int* in_sizes, int n_in,
                              void* d_out, int out_size, void* d_ws,
                              size_t ws_size, hipStream_t stream) {
  const float* x        = (const float*)d_in[0];
  const float* t        = (const float*)d_in[1];
  const float* qkv_w    = (const float*)d_in[2];
  const float* qkv_b    = (const float*)d_in[3];
  const float* proj_w   = (const float*)d_in[4];
  const float* proj_b   = (const float*)d_in[5];
  const float* rel_bias = (const float*)d_in[6];
  const float* fc1_w    = (const float*)d_in[7];
  const float* fc1_b    = (const float*)d_in[8];
  const float* fc2_w    = (const float*)d_in[9];
  const float* fc2_b    = (const float*)d_in[10];
  const float* ada_w1   = (const float*)d_in[11];
  const float* ada_w2   = (const float*)d_in[12];
  const float* ada_b2   = (const float*)d_in[13];
  float* out = (float*)d_out;

  char* ws = (char*)d_ws;
  float*  mod  = (float*)ws;                       // 2*2304 f32
  half_t* wq   = (half_t*)(ws + (64ull << 10));    // f16 weights
  half_t* wp   = wq + 384 * 1152;
  half_t* w1   = wp + 384 * 384;
  half_t* w2   = w1 + 384 * 1536;
  half_t* bufA = (half_t*)(ws + (4ull << 20));     // 48 MiB: xw / attn / h2
  char*   bufB = ws + (68ull << 20);               // <=147 MiB: qkv / proj / mlp1

  // weights -> f16
  f2h_kernel<<<(384 * 1152 + 255) / 256, 256, 0, stream>>>(qkv_w, wq, 384 * 1152);
  f2h_kernel<<<(384 * 384 + 255) / 256, 256, 0, stream>>>(proj_w, wp, 384 * 384);
  f2h_kernel<<<(384 * 1536 + 255) / 256, 256, 0, stream>>>(fc1_w, w1, 384 * 1536);
  f2h_kernel<<<(1536 * 384 + 255) / 256, 256, 0, stream>>>(fc2_w, w2, 1536 * 384);

  // adaLN modulation
  mod_kernel<<<1, 256, 0, stream>>>(t, ada_w1, ada_w2, ada_b2, mod);

  // LN + modulate + shift + partition -> xw (f16, 65536 x 384)
  ln_part_kernel<<<8192, 256, 0, stream>>>(x, mod, bufA);

  // qkv GEMM: 65536 x 1152 = xw @ qkv_w + qkv_b  (f16 out)
  gemm_kernel<0><<<dim3(512, 18), 256, 0, stream>>>(
      bufA, wq, qkv_b, bufB, 65536, 1152, 384, nullptr, nullptr, 1);

  // window attention (512 windows * 12 heads * 2 row-halves)
  attn_kernel<<<12288, 256, 0, stream>>>((const half_t*)bufB, rel_bias, bufA);

  // proj GEMM: 65536 x 384 (f32 out)
  gemm_kernel<2><<<dim3(512, 6), 256, 0, stream>>>(
      bufA, wp, proj_b, bufB, 65536, 384, 384, nullptr, nullptr, 1);

  // reverse roll + gated residual -> x1 (in d_out)
  attn_residual_kernel<<<75264, 256, 0, stream>>>(x, mod, (const float*)bufB, out);

  // LN + modulate for MLP -> h2 (f16, 50176 x 384)
  ln_mlp_kernel<<<6272, 256, 0, stream>>>(out, mod, bufA);

  // fc1 + exact GELU (f16 out, 50176 x 1536)
  gemm_kernel<1><<<dim3(392, 24), 256, 0, stream>>>(
      bufA, w1, fc1_b, bufB, 50176, 1536, 384, nullptr, nullptr, 1);

  // fc2 + fused final residual: out = x1 + g_mlp * (mlp @ fc2_w + fc2_b)
  gemm_kernel<3><<<dim3(392, 6), 256, 0, stream>>>(
      (const half_t*)bufB, w2, fc2_b, out, 50176, 384, 1536, out, mod, 25088);
}
